// VirtualTwin_58583353917986
// MI455X (gfx1250) — compile-verified, matrix operand order checked by reading, not run on hardware
//
#include <hip/hip_runtime.h>
#include <hip/hip_bf16.h>
#include <math.h>

// ---------------------------------------------------------------------------
// RouteNet-style GNN on MI455X (gfx1250, wave32).
// All dense matmuls (path GRU, link attention, link GRU, readout layer 1) are
// mapped to v_wmma_f32_16x16x32_f16: D=32 == K so each 16x16 tile is a single
// WMMA with f32 accumulate. Fragment layouts follow CDNA5 ISA 7.12.2:
//   A (f16 16x32): lane L holds row M=L&15; K = {0..7,16..23} for L<16,
//                  {8..15,24..31} for L>=16 (contiguous pairs per VGPR).
//   B (f16 32x16): lane L holds col N=L&15 with the same K split.
//   C/D (f32 16x16): lane L holds col N=L&15, rows M = r + 8*(L>>4), r=0..7.
// GRU recurrence converts the C-layout h back to an A-layout fragment through
// a per-wave LDS staging tile each step.
// ---------------------------------------------------------------------------

typedef __attribute__((ext_vector_type(16))) _Float16 v16h;
typedef __attribute__((ext_vector_type(8)))  float    v8f;

#define WMMA_F16(a, b, c) \
  __builtin_amdgcn_wmma_f32_16x16x32_f16(false, (a), false, (b), (short)0, (c), false, false)

#define F_N   65536
#define NL_N  8192
#define LP_N  8
#define DG_N  64
#define D_N   32
#define PSS_S 288   // 9 * 32 floats per flow row in pss

__device__ __forceinline__ float selu_f(float x) {
  const float sc = 1.0507009873554805f, al = 1.6732632423543772f;
  return x > 0.f ? sc * x : sc * al * (__expf(x) - 1.f);
}
__device__ __forceinline__ float sigm_f(float x) { return 1.f / (1.f + __expf(-x)); }
__device__ __forceinline__ float softplus_f(float x) { return x > 20.f ? x : log1pf(__expf(x)); }

// ---------------------------------------------------------------------------
// Flow encoder: pf[F,4] -> selu MLP -> path_state[F,32]. Tiny (0.15 GFLOP): scalar.
// ---------------------------------------------------------------------------
__global__ __launch_bounds__(128) void flow_encoder_kernel(
    const float* __restrict__ tr, const float* __restrict__ ln,
    const float* __restrict__ lo, const float* __restrict__ pd,
    const float* __restrict__ w1, const float* __restrict__ b1,
    const float* __restrict__ w2, const float* __restrict__ b2,
    float* __restrict__ path_state) {
  const int f = blockIdx.x * blockDim.x + threadIdx.x;
  const float p0 = (tr[f] - 0.5f) * 2.0f;
  const float p1 = (ln[f] - 0.5f) * 2.0f;
  const float p2 = (lo[f] - 0.1f) * 5.0f;
  const float p3 = (pd[f] - 0.2f) * 3.0f;
  float h[32];
#pragma unroll
  for (int j = 0; j < 32; ++j)
    h[j] = selu_f(b1[j] + p0 * w1[j] + p1 * w1[32 + j] + p2 * w1[64 + j] + p3 * w1[96 + j]);
  for (int j = 0; j < 32; ++j) {
    float a = b2[j];
#pragma unroll
    for (int k = 0; k < 32; ++k) a += h[k] * w2[k * 32 + j];
    path_state[f * 32 + j] = selu_f(a);
  }
}

// ---------------------------------------------------------------------------
// Link encoder: load = mean(traffic[f2l_flow]) / cap; lf[NL,2] -> MLP -> link_state.
// ---------------------------------------------------------------------------
__global__ __launch_bounds__(128) void link_encoder_kernel(
    const float* __restrict__ tr, const float* __restrict__ cap,
    const int* __restrict__ f2l_flow,
    const float* __restrict__ w1, const float* __restrict__ b1,
    const float* __restrict__ w2, const float* __restrict__ b2,
    float* __restrict__ link_state) {
  const int L = blockIdx.x * blockDim.x + threadIdx.x;
  float s = 0.f;
  for (int d = 0; d < DG_N; ++d) s += tr[f2l_flow[L * DG_N + d]];
  const float q0 = (cap[L] - 1.0f) * 1.0f;   // norm link_capacity (1.0, 1.0)
  const float q1 = (s * (1.f / 64.f)) / cap[L];
  float h[32];
#pragma unroll
  for (int j = 0; j < 32; ++j)
    h[j] = selu_f(b1[j] + q0 * w1[j] + q1 * w1[32 + j]);
  for (int j = 0; j < 32; ++j) {
    float a = b2[j];
#pragma unroll
    for (int k = 0; k < 32; ++k) a += h[k] * w2[k * 32 + j];
    link_state[L * 32 + j] = selu_f(a);
  }
}

// ---------------------------------------------------------------------------
// Path GRU: one wave owns 16 flows; 8 sequential steps, 12 WMMAs per step
// (6 col-tiles of [16x32]@[32x96] for x@W and h@U each). Writes pss slots 0..8
// and the final path_state in place.
// ---------------------------------------------------------------------------
__global__ __launch_bounds__(128) void path_gru_kernel(
    float* path_state, const float* __restrict__ link_state,
    const int* __restrict__ link_to_flow,
    const float* __restrict__ pg_w, const float* __restrict__ pg_u,
    const float* __restrict__ pg_b, float* __restrict__ pss) {
  __shared__ float st[4][16][33];  // per-wave C->A relayout staging (pad 33 vs banks)
  const int wave = threadIdx.x >> 5;
  const int lane = threadIdx.x & 31;
  const int flowBase = (blockIdx.x * 4 + wave) * 16;
  const int n  = lane & 15;             // A row / B,C col
  const int k0 = (lane < 16) ? 0 : 8;   // K split half
  const int mC = (lane >> 4) * 8;       // C/D row offset

  // Weight fragments (B layout) + bias values per output column.
  v16h Bw[6], Bu[6];
  float b0v[6], b1v[6];
#pragma unroll
  for (int j = 0; j < 6; ++j) {
#pragma unroll
    for (int t = 0; t < 8; ++t) {
      Bw[j][t]     = (_Float16)pg_w[(k0 + t) * 96 + j * 16 + n];
      Bw[j][t + 8] = (_Float16)pg_w[(k0 + 16 + t) * 96 + j * 16 + n];
      Bu[j][t]     = (_Float16)pg_u[(k0 + t) * 96 + j * 16 + n];
      Bu[j][t + 8] = (_Float16)pg_u[(k0 + 16 + t) * 96 + j * 16 + n];
    }
    b0v[j] = pg_b[j * 16 + n];
    b1v[j] = pg_b[96 + j * 16 + n];
  }

  // h in C layout; also emit pss slot 0 (= prev state).
  v8f hc[2];
#pragma unroll
  for (int d = 0; d < 2; ++d)
#pragma unroll
    for (int r = 0; r < 8; ++r) {
      const float v = path_state[(flowBase + mC + r) * 32 + d * 16 + n];
      hc[d][r] = v;
      pss[(size_t)(flowBase + mC + r) * PSS_S + d * 16 + n] = v;
    }

  for (int s = 0; s < LP_N; ++s) {
    // hc (C layout) -> ha (A layout) through LDS.
    __syncthreads();
#pragma unroll
    for (int d = 0; d < 2; ++d)
#pragma unroll
      for (int r = 0; r < 8; ++r) st[wave][mC + r][d * 16 + n] = hc[d][r];
    __syncthreads();
    v16h ha;
#pragma unroll
    for (int t = 0; t < 8; ++t) {
      ha[t]     = (_Float16)st[wave][n][k0 + t];
      ha[t + 8] = (_Float16)st[wave][n][k0 + 16 + t];
    }
    // x fragment: gathered link state (1 MB table, L2 resident).
    const int link = link_to_flow[(flowBase + n) * LP_N + s];
    v16h xa;
#pragma unroll
    for (int t = 0; t < 8; ++t) {
      xa[t]     = (_Float16)link_state[link * 32 + k0 + t];
      xa[t + 8] = (_Float16)link_state[link * 32 + k0 + 16 + t];
    }
    // mx = x@W + b0, mh = h@U + b1 : 12 WMMAs.
    v8f mx[6], mh[6];
#pragma unroll
    for (int j = 0; j < 6; ++j) {
      v8f cw, cu;
#pragma unroll
      for (int r = 0; r < 8; ++r) { cw[r] = b0v[j]; cu[r] = b1v[j]; }
      mx[j] = WMMA_F16(xa, Bw[j], cw);
      mh[j] = WMMA_F16(ha, Bu[j], cu);
    }
    // Gates elementwise in C layout; write pss slot s+1.
#pragma unroll
    for (int d = 0; d < 2; ++d)
#pragma unroll
      for (int r = 0; r < 8; ++r) {
        const float z  = sigm_f(mx[d][r] + mh[d][r]);
        const float rg = sigm_f(mx[2 + d][r] + mh[2 + d][r]);
        const float c  = tanhf(mx[4 + d][r] + rg * mh[4 + d][r]);
        const float hn = z * hc[d][r] + (1.f - z) * c;
        hc[d][r] = hn;
        pss[(size_t)(flowBase + mC + r) * PSS_S + (s + 1) * 32 + d * 16 + n] = hn;
      }
  }
#pragma unroll
  for (int d = 0; d < 2; ++d)
#pragma unroll
    for (int r = 0; r < 8; ++r)
      path_state[(flowBase + mC + r) * 32 + d * 16 + n] = hc[d][r];
}

// ---------------------------------------------------------------------------
// Link attention: one wave per link. Gather 64 path states from pss, 8 WMMAs
// for attn = leaky_relu([64x32]@[32x32]+b), softmax over D (last axis), then
// agg[n] = sum_rows score*pg. Writes agg[NL,32].
// ---------------------------------------------------------------------------
__global__ __launch_bounds__(64) void link_attn_kernel(
    const float* __restrict__ pss, const int* __restrict__ f2l_flow,
    const int* __restrict__ f2l_pos, const float* __restrict__ att_w,
    const float* __restrict__ att_b, float* __restrict__ agg) {
  __shared__ _Float16 pgh[2][64][32];
  __shared__ float attS[2][64][33];
  __shared__ float rmax[2][64];
  __shared__ float rsum[2][64];
  const int wave = threadIdx.x >> 5;
  const int lane = threadIdx.x & 31;
  const int L  = blockIdx.x * 2 + wave;
  const int n  = lane & 15;
  const int k0 = (lane < 16) ? 0 : 8;
  const int mC = (lane >> 4) * 8;

  // Gather path_gather = pss[f2l_flow, f2l_pos]  (pos in 0..8 selects pss slot).
  for (int rr = lane; rr < DG_N; rr += 32) {
    const int fl = f2l_flow[L * DG_N + rr];
    const int ps = f2l_pos[L * DG_N + rr];
    const float* src = pss + (size_t)fl * PSS_S + ps * 32;
#pragma unroll
    for (int k = 0; k < 32; ++k) pgh[wave][rr][k] = (_Float16)src[k];
  }
  // att_w [32,32] -> 2 B col-tiles.
  v16h Bt[2];
  float bb[2];
#pragma unroll
  for (int d = 0; d < 2; ++d) {
#pragma unroll
    for (int t = 0; t < 8; ++t) {
      Bt[d][t]     = (_Float16)att_w[(k0 + t) * 32 + d * 16 + n];
      Bt[d][t + 8] = (_Float16)att_w[(k0 + 16 + t) * 32 + d * 16 + n];
    }
    bb[d] = att_b[d * 16 + n];
  }
  __syncthreads();
#pragma unroll
  for (int rt = 0; rt < 4; ++rt) {
    v16h A;
#pragma unroll
    for (int t = 0; t < 8; ++t) {
      A[t]     = pgh[wave][rt * 16 + n][k0 + t];
      A[t + 8] = pgh[wave][rt * 16 + n][k0 + 16 + t];
    }
#pragma unroll
    for (int d = 0; d < 2; ++d) {
      v8f c;
#pragma unroll
      for (int r = 0; r < 8; ++r) c[r] = bb[d];
      const v8f o = WMMA_F16(A, Bt[d], c);
#pragma unroll
      for (int r = 0; r < 8; ++r) {
        float v = o[r];
        v = v > 0.f ? v : 0.2f * v;  // leaky_relu slope 0.2
        attS[wave][rt * 16 + mC + r][d * 16 + n] = v;
      }
    }
  }
  __syncthreads();
  // Row-wise softmax stats over D=32 (reference softmax axis=-1).
  for (int rr = lane; rr < DG_N; rr += 32) {
    float m = -1e30f;
    for (int k = 0; k < 32; ++k) m = fmaxf(m, attS[wave][rr][k]);
    float s = 0.f;
    for (int k = 0; k < 32; ++k) s += __expf(attS[wave][rr][k] - m);
    rmax[wave][rr] = m;
    rsum[wave][rr] = s;
  }
  __syncthreads();
  // agg over the 64 gathered paths; lane owns one of the 32 feature columns.
  float acc = 0.f;
  for (int rr = 0; rr < DG_N; ++rr) {
    const float sc = __expf(attS[wave][rr][lane] - rmax[wave][rr]) / rsum[wave][rr];
    acc += sc * (float)pgh[wave][rr][lane];
  }
  agg[L * 32 + lane] = acc;
}

// ---------------------------------------------------------------------------
// Link GRU: single step, one wave per 16 links, 12 WMMAs. In-place link_state.
// ---------------------------------------------------------------------------
__global__ __launch_bounds__(128) void link_gru_kernel(
    float* link_state, const float* __restrict__ agg,
    const float* __restrict__ lg_w, const float* __restrict__ lg_u,
    const float* __restrict__ lg_b) {
  const int wave = threadIdx.x >> 5;
  const int lane = threadIdx.x & 31;
  const int linkBase = (blockIdx.x * 4 + wave) * 16;
  const int n  = lane & 15;
  const int k0 = (lane < 16) ? 0 : 8;
  const int mC = (lane >> 4) * 8;

  v16h Bw[6], Bu[6];
  float b0v[6], b1v[6];
#pragma unroll
  for (int j = 0; j < 6; ++j) {
#pragma unroll
    for (int t = 0; t < 8; ++t) {
      Bw[j][t]     = (_Float16)lg_w[(k0 + t) * 96 + j * 16 + n];
      Bw[j][t + 8] = (_Float16)lg_w[(k0 + 16 + t) * 96 + j * 16 + n];
      Bu[j][t]     = (_Float16)lg_u[(k0 + t) * 96 + j * 16 + n];
      Bu[j][t + 8] = (_Float16)lg_u[(k0 + 16 + t) * 96 + j * 16 + n];
    }
    b0v[j] = lg_b[j * 16 + n];
    b1v[j] = lg_b[96 + j * 16 + n];
  }
  // h in both layouts straight from memory (single step: no relayout needed).
  v8f hc[2];
#pragma unroll
  for (int d = 0; d < 2; ++d)
#pragma unroll
    for (int r = 0; r < 8; ++r)
      hc[d][r] = link_state[(linkBase + mC + r) * 32 + d * 16 + n];
  v16h ha, xa;
#pragma unroll
  for (int t = 0; t < 8; ++t) {
    ha[t]     = (_Float16)link_state[(linkBase + n) * 32 + k0 + t];
    ha[t + 8] = (_Float16)link_state[(linkBase + n) * 32 + k0 + 16 + t];
    xa[t]     = (_Float16)agg[(linkBase + n) * 32 + k0 + t];
    xa[t + 8] = (_Float16)agg[(linkBase + n) * 32 + k0 + 16 + t];
  }
  v8f mx[6], mh[6];
#pragma unroll
  for (int j = 0; j < 6; ++j) {
    v8f cw, cu;
#pragma unroll
    for (int r = 0; r < 8; ++r) { cw[r] = b0v[j]; cu[r] = b1v[j]; }
    mx[j] = WMMA_F16(xa, Bw[j], cw);
    mh[j] = WMMA_F16(ha, Bu[j], cu);
  }
#pragma unroll
  for (int d = 0; d < 2; ++d)
#pragma unroll
    for (int r = 0; r < 8; ++r) {
      const float z  = sigm_f(mx[d][r] + mh[d][r]);
      const float rg = sigm_f(mx[2 + d][r] + mh[2 + d][r]);
      const float c  = tanhf(mx[4 + d][r] + rg * mh[4 + d][r]);
      link_state[(linkBase + mC + r) * 32 + d * 16 + n] =
          z * hc[d][r] + (1.f - z) * c;
    }
}

// ---------------------------------------------------------------------------
// Readout: wave owns 4 flows (= 32 rows of pss[:,1:9,:]). Layer 1 ([32]->16)
// via 2 WMMAs; layers 2/3 scalar per lane-row; deterministic per-flow sum.
// ---------------------------------------------------------------------------
__global__ __launch_bounds__(128) void readout_kernel(
    const float* __restrict__ pss, const int* __restrict__ link_to_flow,
    const float* __restrict__ cap, const float* __restrict__ propag,
    const float* __restrict__ w1, const float* __restrict__ b1,
    const float* __restrict__ w2, const float* __restrict__ b2,
    const float* __restrict__ w3, const float* __restrict__ b3,
    float* __restrict__ out) {
  __shared__ float h1s[4][32][17];
  __shared__ float occ[4][32];
  const int wave = threadIdx.x >> 5;
  const int lane = threadIdx.x & 31;
  const int rowBase = (blockIdx.x * 4 + wave) * 32;  // row = flow*8 + step
  const int n  = lane & 15;
  const int k0 = (lane < 16) ? 0 : 8;
  const int mC = (lane >> 4) * 8;

  v16h B;
#pragma unroll
  for (int t = 0; t < 8; ++t) {
    B[t]     = (_Float16)w1[(k0 + t) * 16 + n];
    B[t + 8] = (_Float16)w1[(k0 + 16 + t) * 16 + n];
  }
  const float bb = b1[n];
#pragma unroll
  for (int rt = 0; rt < 2; ++rt) {
    const int gr = rowBase + rt * 16 + n;
    const float* src = pss + (size_t)(gr >> 3) * PSS_S + ((gr & 7) + 1) * 32;
    v16h A;
#pragma unroll
    for (int t = 0; t < 8; ++t) {
      A[t]     = (_Float16)src[k0 + t];
      A[t + 8] = (_Float16)src[k0 + 16 + t];
    }
    v8f c;
#pragma unroll
    for (int r = 0; r < 8; ++r) c[r] = bb;
    const v8f o = WMMA_F16(A, B, c);
#pragma unroll
    for (int r = 0; r < 8; ++r) h1s[wave][rt * 16 + mC + r][n] = selu_f(o[r]);
  }
  __syncthreads();
  // One lane per row: 16->8->1 scalar tail.
  const int rr = lane;
  const int gr = rowBase + rr;
  const int f = gr >> 3, s = gr & 7;
  float h2[8];
#pragma unroll
  for (int j = 0; j < 8; ++j) {
    float a = b2[j];
#pragma unroll
    for (int k = 0; k < 16; ++k) a += h1s[wave][rr][k] * w2[k * 8 + j];
    h2[j] = selu_f(a);
  }
  float a = b3[0];
#pragma unroll
  for (int j = 0; j < 8; ++j) a += h2[j] * w3[j];
  const int link = link_to_flow[f * LP_N + s];
  occ[wave][rr] = softplus_f(a) / cap[link];
  __syncthreads();
  if ((lane & 7) == 0) {
    float sum = 0.f;
#pragma unroll
    for (int t = 0; t < 8; ++t) sum += occ[wave][rr + t];
    out[f] = sum + propag[f];
  }
}

// ---------------------------------------------------------------------------
extern "C" void kernel_launch(void* const* d_in, const int* in_sizes, int n_in,
                              void* d_out, int out_size, void* d_ws, size_t ws_size,
                              hipStream_t stream) {
  (void)in_sizes; (void)n_in; (void)out_size; (void)ws_size;
  const float* flow_traffic      = (const float*)d_in[0];
  const float* flow_length       = (const float*)d_in[1];
  const float* flow_loss_packet  = (const float*)d_in[2];
  const float* flow_propag_delay = (const float*)d_in[3];
  /* d_in[4] flow_packet_size unused by the reference */
  const float* link_capacity = (const float*)d_in[5];
  const int*   link_to_flow  = (const int*)d_in[6];
  const int*   f2l_flow      = (const int*)d_in[7];
  const int*   f2l_pos       = (const int*)d_in[8];
  const float* fe_w1 = (const float*)d_in[9];
  const float* fe_b1 = (const float*)d_in[10];
  const float* fe_w2 = (const float*)d_in[11];
  const float* fe_b2 = (const float*)d_in[12];
  const float* le_w1 = (const float*)d_in[13];
  const float* le_b1 = (const float*)d_in[14];
  const float* le_w2 = (const float*)d_in[15];
  const float* le_b2 = (const float*)d_in[16];
  const float* att_w = (const float*)d_in[17];
  const float* att_b = (const float*)d_in[18];
  const float* pg_w  = (const float*)d_in[19];
  const float* pg_u  = (const float*)d_in[20];
  const float* pg_b  = (const float*)d_in[21];
  const float* lg_w  = (const float*)d_in[22];
  const float* lg_u  = (const float*)d_in[23];
  const float* lg_b  = (const float*)d_in[24];
  const float* ro_w1 = (const float*)d_in[25];
  const float* ro_b1 = (const float*)d_in[26];
  const float* ro_w2 = (const float*)d_in[27];
  const float* ro_b2 = (const float*)d_in[28];
  const float* ro_w3 = (const float*)d_in[29];
  const float* ro_b3 = (const float*)d_in[30];
  float* out = (float*)d_out;

  // Workspace layout (floats): pss[F*9*32] | path_state[F*32] | link_state[NL*32] | agg[NL*32]
  float* ws         = (float*)d_ws;
  float* pss        = ws;
  float* path_state = pss + (size_t)F_N * PSS_S;
  float* link_state = path_state + (size_t)F_N * D_N;
  float* agg        = link_state + (size_t)NL_N * D_N;

  flow_encoder_kernel<<<F_N / 128, 128, 0, stream>>>(
      flow_traffic, flow_length, flow_loss_packet, flow_propag_delay,
      fe_w1, fe_b1, fe_w2, fe_b2, path_state);
  link_encoder_kernel<<<NL_N / 128, 128, 0, stream>>>(
      flow_traffic, link_capacity, f2l_flow, le_w1, le_b1, le_w2, le_b2, link_state);

  for (int it = 0; it < 8; ++it) {
    path_gru_kernel<<<F_N / 64, 128, 0, stream>>>(
        path_state, link_state, link_to_flow, pg_w, pg_u, pg_b, pss);
    link_attn_kernel<<<NL_N / 2, 64, 0, stream>>>(
        pss, f2l_flow, f2l_pos, att_w, att_b, agg);
    link_gru_kernel<<<NL_N / 64, 128, 0, stream>>>(
        link_state, agg, lg_w, lg_u, lg_b);
  }

  readout_kernel<<<F_N / 16, 128, 0, stream>>>(
      pss, link_to_flow, link_capacity, flow_propag_delay,
      ro_w1, ro_b1, ro_w2, ro_b2, ro_w3, ro_b3, out);
}
